// PhysicsMessageGNN_20512763806108
// MI455X (gfx1250) — matrix operand order repaired and checked by reading, not compiled, not run on hardware
//
#include <hip/hip_runtime.h>
#include <hip/hip_bf16.h>
#include <math.h>

#define NN   4096
#define HID  128
#define INFEATS 64
#define NSLICE 8              // j-dimension split per 16-row block
#define JS (NN / NSLICE)      // 512 columns per wave slice
#define SCALE_INV (1.0f/3.0f)
#define LJ_SIGMA2 12.25f      // 3.5^2

typedef __attribute__((ext_vector_type(16))) _Float16 v16h;
typedef __attribute__((ext_vector_type(8)))  _Float16 v8h;
typedef __attribute__((ext_vector_type(8)))  float    v8f;
typedef __attribute__((ext_vector_type(4)))  float    v4f;

#define WMMA_F16(A,B,C) __builtin_amdgcn_wmma_f32_16x16x32_f16(false,(A),false,(B),(short)0,(C),false,false)

static __device__ __forceinline__ v16h ld16h(const _Float16* p0, const _Float16* p1) {
  v8h lo = *(const v8h*)p0;
  v8h hi = *(const v8h*)p1;
  return __builtin_shufflevector(lo, hi, 0,1,2,3,4,5,6,7,8,9,10,11,12,13,14,15);
}

// ---------------------------------------------------------------------------
// h0 = x @ W_in + b_in ; also emit f16 row-major and f16 transposed copies
// ---------------------------------------------------------------------------
__global__ void k_in(const float* __restrict__ x, const float* __restrict__ W_in,
                     const float* __restrict__ b_in,
                     float* __restrict__ h, _Float16* __restrict__ h16,
                     _Float16* __restrict__ h16T)
{
  int idx = blockIdx.x * blockDim.x + threadIdx.x;
  int row = idx >> 7;
  int col = idx & (HID - 1);
  float a = b_in[col];
#pragma unroll 8
  for (int k = 0; k < INFEATS; ++k)
    a = fmaf(x[(size_t)row * INFEATS + k], W_in[(size_t)k * HID + col], a);
  h[idx] = a;
  _Float16 a16 = (_Float16)a;
  h16[idx] = a16;
  h16T[(size_t)col * NN + row] = a16;
}

// ---------------------------------------------------------------------------
// Transposed f16 copies of W_u1 (256x128 -> 128x256) and W_u2 (128x128)
// ---------------------------------------------------------------------------
__global__ void k_cvtw(const float* __restrict__ W_u1, const float* __restrict__ W_u2,
                       _Float16* __restrict__ wu1t, _Float16* __restrict__ wu2t)
{
  int idx = blockIdx.x * blockDim.x + threadIdx.x;
  if (idx < 2 * HID * HID) {              // W_u1T: (n,k) n in [0,128), k in [0,256)
    int n = idx >> 8, k = idx & 255;
    wu1t[idx] = (_Float16)W_u1[(size_t)k * HID + n];
  } else if (idx < 3 * HID * HID) {       // W_u2T: (n,k) both in [0,128)
    int i2 = idx - 2 * HID * HID;
    int n = i2 >> 7, k = i2 & 127;
    wu2t[i2] = (_Float16)W_u2[(size_t)k * HID + n];
  }
}

// ---------------------------------------------------------------------------
// Fused message kernel. Block = 8 waves = one 16-row block; wave w sweeps
// j in [w*512, (w+1)*512) in chunks of 32:
//   S' = h_J @ h_I^T  (WMMA; C-layout == A-operand layout of the next WMMA)
//   w  = (exp(-d/3)*relu(S') + 0.1*lj) * adj   (pure VALU repack, no shuffle)
//   m_I += w @ h_J    (WMMA, B streamed from transposed f16 copy of h)
// Partials reduced deterministically through LDS.
// ---------------------------------------------------------------------------
__global__ __launch_bounds__(256) void k_msg(
    const _Float16* __restrict__ h16, const _Float16* __restrict__ h16T,
    const float* __restrict__ adj, const float* __restrict__ dist,
    _Float16* __restrict__ m16)
{
  __shared__ __align__(16) float s_part[NSLICE][16 * HID];   // 64 KB
  const int lane = threadIdx.x & 31;
  const int wave = threadIdx.x >> 5;      // 0..7 == j-slice
  const int i0 = blockIdx.x * 16;
  const int r  = lane & 15;
  const int hh = lane >> 4;

  // h_I^T as B-operand: invariant over j, keep in regs (32 VGPRs).
  v16h bI[4];
#pragma unroll
  for (int kc = 0; kc < 4; ++kc) {
    const _Float16* p = h16 + (size_t)(i0 + r) * HID + kc * 32 + hh * 16;
    bI[kc] = ld16h(p, p + 8);
  }

  v8f z = {};
  v8f acc[8];
#pragma unroll
  for (int nt = 0; nt < 8; ++nt) acc[nt] = z;

  const int jbeg = wave * JS;
  for (int j0 = jbeg; j0 < jbeg + JS; j0 += 32) {
    // ---- S'_t = h_Jt(16x128) @ h_I^T(128x16), t=0,1 -----------------------
    v8f s[2]; s[0] = z; s[1] = z;
#pragma unroll
    for (int kc = 0; kc < 4; ++kc) {
#pragma unroll
      for (int t = 0; t < 2; ++t) {
        const _Float16* pa = h16 + (size_t)(j0 + t * 16 + r) * HID + kc * 32;
        v16h a = ld16h(pa + 16 * hh, pa + 16 * hh + 8);
        s[t] = WMMA_F16(a, bI[kc], s[t]);
      }
    }
    // ---- elementwise weight, packed straight into A-operand layout --------
    v8h wlo = {}, whi = {};
#pragma unroll
    for (int t = 0; t < 2; ++t) {
      const size_t eb = (size_t)(i0 + r) * NN + j0 + t * 16 + 8 * hh;
      v4f d0 = *(const v4f*)(dist + eb);
      v4f d1 = *(const v4f*)(dist + eb + 4);
      v4f a0 = *(const v4f*)(adj + eb);
      v4f a1 = *(const v4f*)(adj + eb + 4);
#pragma unroll
      for (int d = 0; d < 8; ++d) {
        float dv = (d < 4) ? d0[d] : d1[d - 4];
        float av = (d < 4) ? a0[d] : a1[d - 4];
        float sv = s[t][d];
        float rr = fmaxf(dv, 1e-6f);
        // q2 = sigma^2 / r^2 via v_rcp_f32 (avoids IEEE divide expansion;
        // dist is in [0.5, 10.5], far from denormal/overflow edges)
        float q2 = LJ_SIGMA2 * __builtin_amdgcn_rcpf(rr * rr);
        float q6 = q2 * q2 * q2;
        float potc = 0.04f * (q6 * q6 - q6);    // 0.1 * 4*0.1 * (sr12 - sr6)
        float decay = __expf(-dv * SCALE_INV);
        float wv = (decay * fmaxf(sv, 0.0f) + potc) * av;
        if (t == 0) wlo[d] = (_Float16)wv;
        else        whi[d] = (_Float16)wv;
      }
    }
    v16h w = __builtin_shufflevector(wlo, whi, 0,1,2,3,4,5,6,7,8,9,10,11,12,13,14,15);
    // ---- m_I += w(16x32) @ h_J(32x128) ------------------------------------
#pragma unroll
    for (int nt = 0; nt < 8; ++nt) {
      const _Float16* pb = h16T + (size_t)(nt * 16 + r) * NN + j0;
      v16h b = ld16h(pb + 8 * hh, pb + 16 + 8 * hh);
      acc[nt] = WMMA_F16(w, b, acc[nt]);
    }
  }

  // ---- stage per-wave partial (C layout: vgpr d -> row d+8hh, col nt*16+r)
#pragma unroll
  for (int nt = 0; nt < 8; ++nt)
#pragma unroll
    for (int d = 0; d < 8; ++d)
      s_part[wave][(d + 8 * hh) * HID + nt * 16 + r] = acc[nt][d];
  __syncthreads();

  // ---- deterministic reduction over the 8 slices, emit f16 message -------
  const int tid = threadIdx.x;
#pragma unroll
  for (int e = 0; e < 8; ++e) {
    int idx = tid * 8 + e;                 // 0..2047 over 16x128 tile
    float v = 0.0f;
#pragma unroll
    for (int w2 = 0; w2 < NSLICE; ++w2) v += s_part[w2][idx];
    m16[(size_t)(i0 + (idx >> 7)) * HID + (idx & (HID - 1))] = (_Float16)v;
  }
}

// ---------------------------------------------------------------------------
// Update: t = relu([h|m]@W_u1 + b_u1); delta = t@W_u2 + b_u2;
//         h_new = LayerNorm(h + delta)*gamma + beta  (+ f16 copies)
// ---------------------------------------------------------------------------
__global__ __launch_bounds__(128) void k_update(
    const float* hin, const _Float16* __restrict__ h16in,
    const _Float16* __restrict__ m16,
    const _Float16* __restrict__ wu1t, const _Float16* __restrict__ wu2t,
    const float* __restrict__ b_u1, const float* __restrict__ b_u2,
    const float* __restrict__ gamma, const float* __restrict__ beta,
    float* hout, _Float16* h16out, _Float16* h16Tout)
{
  __shared__ __align__(16) _Float16 s_t16[4][16 * HID];
  __shared__ __align__(16) float    s_del[4][16 * HID];
  const int lane = threadIdx.x & 31;
  const int wave = threadIdx.x >> 5;
  const int i0 = (blockIdx.x * 4 + wave) * 16;
  const int r  = lane & 15;
  const int hh = lane >> 4;
  v8f z = {};

  v8f acc[8];
#pragma unroll
  for (int nt = 0; nt < 8; ++nt) acc[nt] = z;
#pragma unroll
  for (int kc = 0; kc < 8; ++kc) {
    const _Float16* src = (kc < 4)
        ? (h16in + (size_t)(i0 + r) * HID + kc * 32)
        : (m16   + (size_t)(i0 + r) * HID + (kc - 4) * 32);
    v16h a = ld16h(src + 16 * hh, src + 16 * hh + 8);
#pragma unroll
    for (int nt = 0; nt < 8; ++nt) {
      const _Float16* pb = wu1t + (size_t)(nt * 16 + r) * (2 * HID) + kc * 32 + 16 * hh;
      v16h b = ld16h(pb, pb + 8);
      acc[nt] = WMMA_F16(a, b, acc[nt]);
    }
  }
#pragma unroll
  for (int nt = 0; nt < 8; ++nt) {
    float bv = b_u1[nt * 16 + r];
#pragma unroll
    for (int d = 0; d < 8; ++d) {
      float v = acc[nt][d] + bv;
      s_t16[wave][(d + 8 * hh) * HID + nt * 16 + r] = (_Float16)fmaxf(v, 0.0f);
    }
  }
  __syncthreads();

  v8f acc2[8];
#pragma unroll
  for (int nt = 0; nt < 8; ++nt) acc2[nt] = z;
#pragma unroll
  for (int kc = 0; kc < 4; ++kc) {
    const _Float16* pa = &s_t16[wave][r * HID + kc * 32];
    v16h a = ld16h(pa + 16 * hh, pa + 16 * hh + 8);
#pragma unroll
    for (int nt = 0; nt < 8; ++nt) {
      const _Float16* pb = wu2t + (size_t)(nt * 16 + r) * HID + kc * 32 + 16 * hh;
      v16h b = ld16h(pb, pb + 8);
      acc2[nt] = WMMA_F16(a, b, acc2[nt]);
    }
  }
#pragma unroll
  for (int nt = 0; nt < 8; ++nt) {
    float bv = b_u2[nt * 16 + r];
#pragma unroll
    for (int d = 0; d < 8; ++d)
      s_del[wave][(d + 8 * hh) * HID + nt * 16 + r] = acc2[nt][d] + bv;
  }
  __syncthreads();

  if (lane < 16) {
    const int row = i0 + lane;
    const float* hrow = hin + (size_t)row * HID;
    const float* drow = &s_del[wave][lane * HID];
    float sum = 0.0f, sumsq = 0.0f;
    for (int c = 0; c < HID; ++c) {
      float v = hrow[c] + drow[c];
      sum += v; sumsq += v * v;
    }
    float mu  = sum * (1.0f / HID);
    float var = sumsq * (1.0f / HID) - mu * mu;
    float rinv = rsqrtf(var + 1e-5f);
    for (int c = 0; c < HID; ++c) {
      float v = (hrow[c] + drow[c] - mu) * rinv * gamma[c] + beta[c];
      hout[(size_t)row * HID + c] = v;
      _Float16 v16 = (_Float16)v;
      h16out[(size_t)row * HID + c] = v16;
      h16Tout[(size_t)c * NN + row] = v16;
    }
  }
}

// ---------------------------------------------------------------------------
extern "C" void kernel_launch(void* const* d_in, const int* in_sizes, int n_in,
                              void* d_out, int out_size, void* d_ws, size_t ws_size,
                              hipStream_t stream) {
  (void)in_sizes; (void)n_in; (void)out_size; (void)ws_size;
  const float* x     = (const float*)d_in[0];
  const float* adj   = (const float*)d_in[1];
  const float* dist  = (const float*)d_in[2];
  const float* W_in  = (const float*)d_in[3];
  const float* b_in  = (const float*)d_in[4];
  // d_in[5]=W_msg, d_in[6]=b_msg: dead code in the reference
  const float* W_u1  = (const float*)d_in[7];
  const float* b_u1  = (const float*)d_in[8];
  const float* W_u2  = (const float*)d_in[9];
  const float* b_u2  = (const float*)d_in[10];
  const float* gamma = (const float*)d_in[11];
  const float* beta  = (const float*)d_in[12];

  char* ws = (char*)d_ws;
  float*    h    = (float*)   (ws);                            // 2 MB
  _Float16* h16  = (_Float16*)(ws + (2u << 20));               // 1 MB
  _Float16* h16T = (_Float16*)(ws + (3u << 20));               // 1 MB
  _Float16* m16  = (_Float16*)(ws + (4u << 20));               // 1 MB
  _Float16* wu1t = (_Float16*)(ws + (5u << 20));               // 64 KB
  _Float16* wu2t = (_Float16*)(ws + (5u << 20) + (64u << 10)); // 32 KB

  k_in  <<<(NN * HID) / 256, 256, 0, stream>>>(x, W_in, b_in, h, h16, h16T);
  k_cvtw<<<(3 * HID * HID + 255) / 256, 256, 0, stream>>>(W_u1, W_u2, wu1t, wu2t);

  for (int s = 0; s < 3; ++s) {
    k_msg<<<NN / 16, 256, 0, stream>>>(h16, h16T, adj, dist, m16);
    float* hout = (s == 2) ? (float*)d_out : h;
    k_update<<<NN / 64, 128, 0, stream>>>(h, h16, m16, wu1t, wu2t,
                                          b_u1, b_u2, gamma, beta,
                                          hout, h16, h16T);
  }
}
